// CTRGC_35588099015452
// MI455X (gfx1250) — matrix operand order repaired and checked by reading, not compiled, not run on hardware
//
#include <hip/hip_runtime.h>
#include <hip/hip_bf16.h>
#include <math.h>

#define DEV static __device__ __forceinline__

typedef __attribute__((ext_vector_type(8)))  float  v8f;
typedef __attribute__((ext_vector_type(16))) __bf16 v16bf;

// ---- problem constants ----
constexpr int Nn = 16, Cc = 64, Tt = 256, Vv = 25, O4 = 32, Kk = 5;
constexpr int NC      = Nn * Cc;       // 1024
constexpr int SLABS   = Nn * O4;       // 512  (n,c) slabs
constexpr int VP      = 32;            // V padded to 32 for K=32 WMMA
constexpr int TV      = Tt * Vv;       // 6400
constexpr int SLAB_BF = Tt * VP;       // 8192 bf16 elems per padded slab

// ---- bf16 helpers (explicit RNE, no reliance on libdevice conversions) ----
DEV __bf16 f2bf(float f) {
  unsigned u = __float_as_uint(f);
  u += 0x7FFFu + ((u >> 16) & 1u);
  unsigned short s = (unsigned short)(u >> 16);
  __bf16 r; __builtin_memcpy(&r, &s, 2);
  return r;
}

// ---- WMMA fragment loaders (layouts per CDNA5 ISA §7.12.2) ----
// A (16x32 bf16, M rows, K contiguous): lane h=lane/16, m=lane%16 holds
// K=[h*8, h*8+8) then K=[16+h*8, 16+h*8+8)  -> two contiguous 16B chunks.
DEV v16bf load_a16x32(const __bf16* p, int ld) {
  int lane = threadIdx.x & 31;
  int m = lane & 15, h = lane >> 4;
  const __bf16* row = p + m * ld;
  v16bf a;
  __builtin_memcpy(&a, row + h * 8, 16);
  __builtin_memcpy((char*)&a + 16, row + 16 + h * 8, 16);
  return a;
}
// B (32x16): stored as B^T rows [n][k]; lane n=lane%16 holds K=[h*8,h*8+8)
// then K=[16+h*8, 16+h*8+8) -> two contiguous 16B chunks.
DEV v16bf load_bt16x32(const __bf16* p, int ld) {
  int lane = threadIdx.x & 31;
  int n = lane & 15, h = lane >> 4;
  v16bf b;
  __builtin_memcpy(&b, p + n * ld + h * 8, 16);
  __builtin_memcpy((char*)&b + 16, p + n * ld + 16 + h * 8, 16);
  return b;
}
DEV v8f wmma_bf(v16bf a, v16bf b, v8f c) {
  return __builtin_amdgcn_wmma_f32_16x16x32_bf16(false, a, false, b, (short)0, c,
                                                 false, false);
}
// D tile f32 16x16: vgpr r, lane: m = r + 8*(lane/16), n = lane%16

// =====================  graph build  =====================

__global__ void k_mean(const float* __restrict__ x, float* __restrict__ X,
                       __bf16* __restrict__ Xbf) {
  int id = blockIdx.x * blockDim.x + threadIdx.x;   // NC*T threads
  int nc = id & (NC - 1);
  int t  = id >> 10;
  const float* px = x + ((long)nc * Tt + t) * Vv;
  float s = 0.f;
#pragma unroll
  for (int v = 0; v < Vv; ++v) s += px[v];
  s *= (1.0f / Vv);
  X[t * NC + nc]   = s;
  Xbf[t * NC + nc] = f2bf(s);
}

__global__ void k_rownorm(const float* __restrict__ X, float* __restrict__ sq) {
  int t = threadIdx.x;
  float s = 0.f;
  for (int k = 0; k < NC; ++k) { float v = X[t * NC + k]; s += v * v; }
  sq[t] = s;
}

__global__ void k_gram(const __bf16* __restrict__ Xbf, const float* __restrict__ sq,
                       float* __restrict__ D) {
  int wave = blockIdx.x * (blockDim.x >> 5) + (threadIdx.x >> 5); // 256 waves
  int ti = wave >> 4, tj = wave & 15;
  v8f acc = {0.f,0.f,0.f,0.f,0.f,0.f,0.f,0.f};
  for (int kk = 0; kk < NC; kk += 32) {
    v16bf a = load_a16x32(Xbf + ti * 16 * NC + kk, NC);
    v16bf b = load_bt16x32(Xbf + tj * 16 * NC + kk, NC);
    acc = wmma_bf(a, b, acc);
  }
  int lane = threadIdx.x & 31;
  int n = lane & 15, mh = (lane >> 4) * 8;
#pragma unroll
  for (int r = 0; r < 8; ++r) {
    int i = ti * 16 + mh + r, j = tj * 16 + n;
    float d = sqrtf(fmaxf(sq[i] + sq[j] - 2.0f * acc[r], 0.f));
    if (i == j) d = 0.f;
    D[i * Tt + j] = d;
  }
}

__global__ void k_avg(const float* __restrict__ D, float* __restrict__ avgp) {
  __shared__ float red[256];
  float s = 0.f;
  for (int i = threadIdx.x; i < Tt * Tt; i += 256) s += D[i];
  red[threadIdx.x] = s;
  __syncthreads();
  for (int st = 128; st > 0; st >>= 1) {
    if (threadIdx.x < st) red[threadIdx.x] += red[threadIdx.x + st];
    __syncthreads();
  }
  if (threadIdx.x == 0) avgp[0] = red[0] * (1.0f / (Tt * Tt));
}

__global__ void k_topk_h(const float* __restrict__ D, const float* __restrict__ avgp,
                         float* __restrict__ H) {
  int t = threadIdx.x;
  for (int e = 0; e < Tt; ++e) H[t * Tt + e] = 0.f;
  __syncthreads();
  float bv[Kk]; int bi[Kk];
#pragma unroll
  for (int k = 0; k < Kk; ++k) { bv[k] = 3.4e38f; bi[k] = 0; }
  for (int j = 0; j < Tt; ++j) {         // ascending j => stable tie-break
    float d = D[t * Tt + j];
    if (d < bv[Kk - 1]) {
      int p = Kk - 1;
      while (p > 0 && d < bv[p - 1]) { bv[p] = bv[p - 1]; bi[p] = bi[p - 1]; --p; }
      bv[p] = d; bi[p] = j;
    }
  }
  float avg = avgp[0];
  float inv = 1.0f / (avg * avg);
#pragma unroll
  for (int k = 0; k < Kk; ++k) {         // thread t owns column t -> no races
    float val = expf(-(bv[k] * bv[k]) * inv);
    H[bi[k] * Tt + t] += val;
  }
}

__global__ void k_deg(const float* __restrict__ H, float* __restrict__ DVv,
                      float* __restrict__ DEv) {
  int t = threadIdx.x;
  float r = 0.f, c = 0.f;
  for (int e = 0; e < Tt; ++e) { r += H[t * Tt + e]; c += H[e * Tt + t]; }
  DVv[t] = r; DEv[t] = c;
}

__global__ void k_G(const float* __restrict__ H, const float* __restrict__ DVv,
                    const float* __restrict__ DEv, float* __restrict__ T1f,
                    __bf16* __restrict__ T1t) {
  int i = blockIdx.x, j = threadIdx.x;
  __shared__ float Hrow[256], Hcol[256], iDE[256], iDV[256];
  Hrow[j] = H[i * Tt + j];
  Hcol[j] = H[j * Tt + i];
  iDE[j]  = 1.0f / DEv[j];
  iDV[j]  = 1.0f / DVv[j];
  __syncthreads();
  float a1 = 0.f, a2 = 0.f;
  for (int e = 0; e < Tt; ++e) {
    a1 += Hrow[e] * H[j * Tt + e] * iDE[e];   // G(H)
    a2 += Hcol[e] * H[e * Tt + j] * iDV[e];   // G(H^T)
  }
  float g = rsqrtf(DVv[i]) * rsqrtf(DVv[j]) * a1 +
            rsqrtf(DEv[i]) * rsqrtf(DEv[j]) * a2;
  float tv = truncf(g);
  T1f[i * Tt + j] = tv;
  T1t[j * Tt + i] = f2bf(tv);                 // T1^T, bf16, A-operand for x2
}

__global__ void k_vt(const float* __restrict__ T1f, const float* __restrict__ A,
                     const float* __restrict__ A5, __bf16* __restrict__ vt1b,
                     __bf16* __restrict__ vt2b, __bf16* __restrict__ Ab,
                     __bf16* __restrict__ A5b) {
  __shared__ float Acm[VP], Arm[VP], cm[256];
  int t = threadIdx.x;
  float rm = 0.f, cmv = 0.f;
  for (int j = 0; j < Tt; ++j) { rm += T1f[t * Tt + j]; cmv += T1f[j * Tt + t]; }
  rm *= (1.f / 256.f); cmv *= (1.f / 256.f);
  cm[t] = cmv;
  if (t < VP) {
    float ac = 0.f, ar = 0.f;
    if (t < Vv) {
      for (int u = 0; u < Vv; ++u) { ac += A[u * Vv + t]; ar += A[t * Vv + u]; }
      ac *= (1.f / 25.f); ar *= (1.f / 25.f);
    }
    Acm[t] = ac; Arm[t] = ar;
    for (int u = 0; u < VP; ++u) {        // padded bf16 copies of A / A5 (B^T)
      Ab [t * VP + u] = f2bf((t < Vv && u < Vv) ? A [t * Vv + u] : 0.f);
      A5b[t * VP + u] = f2bf((t < Vv && u < Vv) ? A5[t * Vv + u] : 0.f);
    }
  }
  __syncthreads();
  for (int v = 0; v < VP; ++v)            // vt1 [T][VP]  (B^T for x3)
    vt1b[t * VP + v] = f2bf(v < Vv ? tanhf(Acm[v] - rm) : 0.f);
  for (int u = 0; u < VP; ++u)            // vt2 [VP][T]  (B^T for x4)
    vt2b[u * Tt + t] = f2bf(u < Vv ? tanhf(cm[t] - Arm[u]) : 0.f);
}

// =====================  1x1 convs -> padded bf16 operand buffers  ============

__global__ void k_conv(const float* __restrict__ x,
                       const float* __restrict__ w3,  const float* __restrict__ b3,
                       const float* __restrict__ w33, const float* __restrict__ b33,
                       const float* __restrict__ w31, const float* __restrict__ b31,
                       const float* __restrict__ w32, const float* __restrict__ b32,
                       __bf16* __restrict__ c3b,  __bf16* __restrict__ c3t,
                       __bf16* __restrict__ c33b, __bf16* __restrict__ c31b,
                       __bf16* __restrict__ c32t) {
  long id = (long)blockIdx.x * blockDim.x + threadIdx.x; // 4*512*256*32
  int v    = (int)(id & 31);
  int t    = (int)((id >> 5) & 255);
  int slab = (int)((id >> 13) & 511);
  int s    = (int)(id >> 22);
  int o = slab & 31, n = slab >> 5;
  const float* w; const float* b;
  if (s == 0)      { w = w3;  b = b3;  }
  else if (s == 1) { w = w33; b = b33; }
  else if (s == 2) { w = w31; b = b31; }
  else             { w = w32; b = b32; }
  float acc = 0.f;
  if (v < Vv) {
    acc = b[o];
    const float* px = x + ((long)n * Cc * Tt + t) * Vv + v;
    const float* pw = w + o * Cc;
    for (int c = 0; c < Cc; ++c) acc += pw[c] * px[(long)c * Tt * Vv];
  }
  __bf16 bfv = f2bf(acc);                     // f2bf(0)==0 -> pads zeroed
  long sb = (long)slab * SLAB_BF;
  if (s == 0)      { c3b[sb + t * VP + v] = bfv; c3t[sb + v * Tt + t] = bfv; }
  else if (s == 1) { c33b[sb + t * VP + v] = bfv; }
  else if (s == 2) { c31b[sb + t * VP + v] = bfv; }
  else             { c32t[sb + v * Tt + t] = bfv; }
}

// ====================  x1 = A*c3 + alpha*T1^T*c3 ; x11 = (x1 + A5*c33)/2 =====

__global__ void k_x1(const __bf16* __restrict__ c3b, const __bf16* __restrict__ c3t,
                     const __bf16* __restrict__ c33b, const __bf16* __restrict__ Ab,
                     const __bf16* __restrict__ A5b, const __bf16* __restrict__ T1t,
                     const int* __restrict__ alphap, float* __restrict__ x1f,
                     __bf16* __restrict__ x1t, float* __restrict__ x11f,
                     __bf16* __restrict__ x11b) {
  int slab = blockIdx.x;
  int wave = threadIdx.x >> 5, lane = threadIdx.x & 31;
  int nn = lane & 15, mh = (lane >> 4) * 8;
  float alpha = (float)alphap[0];
  long sb = (long)slab * SLAB_BF;
  v8f z = {0.f,0.f,0.f,0.f,0.f,0.f,0.f,0.f};
  for (int mi = 0; mi < 2; ++mi) {
    int t0 = (wave + mi * 8) * 16;
    v16bf a_c3  = load_a16x32(c3b  + sb + t0 * VP, VP);
    v16bf a_c33 = load_a16x32(c33b + sb + t0 * VP, VP);
    for (int nt = 0; nt < 2; ++nt) {
      v8f acc1 = wmma_bf(a_c3, load_bt16x32(Ab + nt * 16 * VP, VP), z);
      v8f acc2 = z;
      for (int kk = 0; kk < Tt; kk += 32) {
        v16bf a = load_a16x32(T1t + t0 * Tt + kk, Tt);
        v16bf b = load_bt16x32(c3t + sb + nt * 16 * Tt + kk, Tt);
        acc2 = wmma_bf(a, b, acc2);
      }
      v8f acc3 = wmma_bf(a_c33, load_bt16x32(A5b + nt * 16 * VP, VP), z);
#pragma unroll
      for (int r = 0; r < 8; ++r) {
        int t = t0 + mh + r, v = nt * 16 + nn;
        float xv   = acc1[r] + alpha * acc2[r];
        float x11v = (xv + acc3[r]) * 0.5f;
        if (v < Vv) {
          x1f [(long)slab * TV + t * Vv + v] = xv;
          x11f[(long)slab * TV + t * Vv + v] = x11v;
        } else { xv = 0.f; x11v = 0.f; }
        x1t [sb + v * Tt + t] = f2bf(xv);     // B^T for x5
        x11b[sb + t * VP + v] = f2bf(x11v);   // A for x6
      }
    }
  }
}

// ====================  x4[v,u] = sum_t c32[t,v]*vt2[u,t]  ====================

__global__ void k_x4(const __bf16* __restrict__ c32t, const __bf16* __restrict__ vt2b,
                     __bf16* __restrict__ x4b) {
  int slab = blockIdx.x;
  int mv = threadIdx.x >> 5, lane = threadIdx.x & 31;  // 2 waves = 2 v-tiles
  int nn = lane & 15, mh = (lane >> 4) * 8;
  long sb = (long)slab * SLAB_BF;
  v8f z = {0.f,0.f,0.f,0.f,0.f,0.f,0.f,0.f};
  for (int nu = 0; nu < 2; ++nu) {
    v8f acc = z;
    for (int kk = 0; kk < Tt; kk += 32) {
      v16bf a = load_a16x32(c32t + sb + mv * 16 * Tt + kk, Tt);
      v16bf b = load_bt16x32(vt2b + nu * 16 * Tt + kk, Tt);
      acc = wmma_bf(a, b, acc);
    }
#pragma unroll
    for (int r = 0; r < 8; ++r) {
      int v = mv * 16 + mh + r, u = nu * 16 + nn;
      float val = (v < Vv && u < Vv) ? acc[r] : 0.f;
      x4b[(long)slab * (VP * VP) + v * VP + u] = f2bf(val); // B^T for x6
    }
  }
}

// ======  x3 = c31*vt1^T (stays in 128KB LDS), x5 = x3*x1  (per slab) ========

__global__ void k_x3x5(const __bf16* __restrict__ c31b, const __bf16* __restrict__ vt1b,
                       const __bf16* __restrict__ x1t, float* __restrict__ x5f) {
  extern __shared__ __bf16 x3s[];                 // 256*256 bf16 = 128 KB
  int slab = blockIdx.x;
  int wave = threadIdx.x >> 5, lane = threadIdx.x & 31;
  int nn = lane & 15, mh = (lane >> 4) * 8;
  long sb = (long)slab * SLAB_BF;
  v8f z = {0.f,0.f,0.f,0.f,0.f,0.f,0.f,0.f};
  // phase 1: 256 tiles of x3 into LDS
  for (int i = 0; i < 32; ++i) {
    int tt = wave + i * 8;
    int mt = tt >> 4, nt = tt & 15;
    v16bf a = load_a16x32(c31b + sb + mt * 16 * VP, VP);
    v16bf b = load_bt16x32(vt1b + nt * 16 * VP, VP);
    v8f acc = wmma_bf(a, b, z);
#pragma unroll
    for (int r = 0; r < 8; ++r)
      x3s[(mt * 16 + mh + r) * Tt + nt * 16 + nn] = f2bf(acc[r]);
  }
  __syncthreads();
  // phase 2: x5 = x3 (LDS, A-operand) * x1 (x1t, B^T-operand)
  for (int i = 0; i < 4; ++i) {
    int tt = wave * 4 + i;
    int mt = tt >> 1, nv = tt & 1;
    v8f acc = z;
    for (int kk = 0; kk < Tt; kk += 32) {
      v16bf a = load_a16x32(x3s + mt * 16 * Tt + kk, Tt);
      v16bf b = load_bt16x32(x1t + sb + nv * 16 * Tt + kk, Tt);
      acc = wmma_bf(a, b, acc);
    }
#pragma unroll
    for (int r = 0; r < 8; ++r) {
      int t = mt * 16 + mh + r, v = nv * 16 + nn;
      if (v < Vv) x5f[(long)slab * TV + t * Vv + v] = acc[r];
    }
  }
}

// ============  x6 = x11*x4^T ; out = concat(x7, x5, x1, x11)  ================

__global__ void k_final(const __bf16* __restrict__ x11b, const __bf16* __restrict__ x4b,
                        const float* __restrict__ x5f, const float* __restrict__ x1f,
                        const float* __restrict__ x11f, float* __restrict__ out) {
  int slab = blockIdx.x;
  int wave = threadIdx.x >> 5, lane = threadIdx.x & 31;
  int nn = lane & 15, mh = (lane >> 4) * 8;
  long sb = (long)slab * SLAB_BF;
  int n = slab >> 5, ch = slab & 31;
  v8f z = {0.f,0.f,0.f,0.f,0.f,0.f,0.f,0.f};
  for (int i = 0; i < 4; ++i) {
    int tt = wave * 4 + i;
    int mt = tt >> 1, nv = tt & 1;
    v16bf a = load_a16x32(x11b + sb + mt * 16 * VP, VP);
    v16bf b = load_bt16x32(x4b + (long)slab * (VP * VP) + nv * 16 * VP, VP);
    v8f x6t = wmma_bf(a, b, z);
#pragma unroll
    for (int r = 0; r < 8; ++r) {
      int t = mt * 16 + mh + r, v = nv * 16 + nn;
      if (v < Vv) {
        long si = (long)slab * TV + t * Vv + v;
        float x5v = x5f[si];
        long base = (((long)n * 128 + ch) * Tt + t) * Vv + v;
        out[base]                   = (x5v + x6t[r]) * 0.5f;   // x7
        out[base + 32L * TV]        = x5v;                     // x5
        out[base + 64L * TV]        = x1f[si];                 // x1
        out[base + 96L * TV]        = x11f[si];                // x11
      }
    }
  }
}

// =============================  host glue  ===================================

extern "C" void kernel_launch(void* const* d_in, const int* in_sizes, int n_in,
                              void* d_out, int out_size, void* d_ws, size_t ws_size,
                              hipStream_t stream) {
  (void)in_sizes; (void)n_in; (void)out_size; (void)ws_size;
  const float* x   = (const float*)d_in[0];
  const float* A   = (const float*)d_in[1];
  const float* A5  = (const float*)d_in[2];
  const int*   alp = (const int*)d_in[3];
  const float* w3  = (const float*)d_in[4];
  const float* b3  = (const float*)d_in[5];
  const float* w33 = (const float*)d_in[6];
  const float* b33 = (const float*)d_in[7];
  const float* w31 = (const float*)d_in[8];
  const float* b31 = (const float*)d_in[9];
  const float* w32 = (const float*)d_in[10];
  const float* b32 = (const float*)d_in[11];
  float* out = (float*)d_out;

  char* ws = (char*)d_ws;
  size_t off = 0;
  auto alloc = [&](size_t bytes) -> char* {
    char* p = ws + off;
    off = (off + bytes + 255) & ~(size_t)255;
    return p;
  };
  float*  X    = (float*) alloc((size_t)Tt * NC * 4);
  __bf16* Xbf  = (__bf16*)alloc((size_t)Tt * NC * 2);
  float*  sq   = (float*) alloc(Tt * 4);
  float*  D    = (float*) alloc((size_t)Tt * Tt * 4);
  float*  avgp = (float*) alloc(256);
  float*  H    = (float*) alloc((size_t)Tt * Tt * 4);
  float*  DVv  = (float*) alloc(Tt * 4);
  float*  DEv  = (float*) alloc(Tt * 4);
  float*  T1f  = (float*) alloc((size_t)Tt * Tt * 4);
  __bf16* T1t  = (__bf16*)alloc((size_t)Tt * Tt * 2);
  __bf16* vt1b = (__bf16*)alloc((size_t)Tt * VP * 2);
  __bf16* vt2b = (__bf16*)alloc((size_t)VP * Tt * 2);
  __bf16* Ab   = (__bf16*)alloc((size_t)VP * VP * 2);
  __bf16* A5b  = (__bf16*)alloc((size_t)VP * VP * 2);
  __bf16* c3b  = (__bf16*)alloc((size_t)SLABS * SLAB_BF * 2);
  __bf16* c3t  = (__bf16*)alloc((size_t)SLABS * SLAB_BF * 2);
  __bf16* c33b = (__bf16*)alloc((size_t)SLABS * SLAB_BF * 2);
  __bf16* c31b = (__bf16*)alloc((size_t)SLABS * SLAB_BF * 2);
  __bf16* c32t = (__bf16*)alloc((size_t)SLABS * SLAB_BF * 2);
  float*  x1f  = (float*) alloc((size_t)SLABS * TV * 4);
  __bf16* x1t  = (__bf16*)alloc((size_t)SLABS * SLAB_BF * 2);
  float*  x11f = (float*) alloc((size_t)SLABS * TV * 4);
  __bf16* x11b = (__bf16*)alloc((size_t)SLABS * SLAB_BF * 2);
  __bf16* x4b  = (__bf16*)alloc((size_t)SLABS * VP * VP * 2);
  float*  x5f  = (float*) alloc((size_t)SLABS * TV * 4);

  // graph build
  k_mean<<<(NC * Tt) / 256, 256, 0, stream>>>(x, X, Xbf);
  k_rownorm<<<1, 256, 0, stream>>>(X, sq);
  k_gram<<<32, 256, 0, stream>>>(Xbf, sq, D);
  k_avg<<<1, 256, 0, stream>>>(D, avgp);
  k_topk_h<<<1, 256, 0, stream>>>(D, avgp, H);
  k_deg<<<1, 256, 0, stream>>>(H, DVv, DEv);
  k_G<<<Tt, 256, 0, stream>>>(H, DVv, DEv, T1f, T1t);
  k_vt<<<1, 256, 0, stream>>>(T1f, A, A5, vt1b, vt2b, Ab, A5b);
  // convs
  k_conv<<<(4L * SLABS * Tt * VP) / 256, 256, 0, stream>>>(
      x, w3, b3, w33, b33, w31, b31, w32, b32, c3b, c3t, c33b, c31b, c32t);
  // einsum chain
  k_x1<<<SLABS, 256, 0, stream>>>(c3b, c3t, c33b, Ab, A5b, T1t, alp,
                                  x1f, x1t, x11f, x11b);
  k_x4<<<SLABS, 64, 0, stream>>>(c32t, vt2b, x4b);
  k_x3x5<<<SLABS, 256, (size_t)Tt * Tt * 2, stream>>>(c31b, vt1b, x1t, x5f);
  k_final<<<SLABS, 256, 0, stream>>>(x11b, x4b, x5f, x1f, x11f, out);
}
